// PCTVQVAE_MASK_57097295233308
// MI455X (gfx1250) — matrix operand order repaired and checked by reading, not compiled, not run on hardware
//
#include <hip/hip_runtime.h>
#include <hip/hip_bf16.h>

// ---------------------------------------------------------------------------
// PCT VQ-VAE forward for MI455X (gfx1250, wave32, WMMA).
// One workgroup per batch sample; residual stream lives in LDS; every 512-deep
// contraction AND the token-mixing MLP run on v_wmma_f32_16x16x32_bf16 with
// bf16 weights streamed from L2 in B-fragment-native transposed layouts
// (forced global_load_b128 via address_space(1)). VQ argmin fused into the
// codebook-GEMM epilogue via packed u64 ds_min atomics.
// ---------------------------------------------------------------------------

#define BB   2048
#define JJ   23
#define JP   32
#define HH   512
#define TIc  64
#define LL   4
#define DD   512
#define KK   2048
#define NT_H 32    // 512/16 N-tiles
#define NT_K 128   // 2048/16 N-tiles
#define THREADS 256
#define ATS  40    // padded row stride (bf16 elems) of transposed A buffer

typedef unsigned short u16;
typedef unsigned int   u32;
typedef unsigned long long u64;
typedef __attribute__((ext_vector_type(16))) __bf16 v16bf;
typedef __attribute__((ext_vector_type(8)))  float  v8f;
typedef __attribute__((ext_vector_type(4)))  unsigned int u32x4;

// LDS layout (bytes)
#define LDS_XR    0        // [JP][512] f32                         (65536)
#define LDS_A0    65536    // [JP][512] bf16                        (32768)
#define LDS_A1    98304    // [JP][512] bf16                        (32768)
                           // T1 = A0..A1 region: [512][64] bf16    (65536)
#define LDS_AT    131072   // [512][ATS] bf16 transposed A          (40960)
#define LDS_MEAN  172032
#define LDS_RSTD  172160
#define LDS_BEST  172288
#define LDS_LOSS  172544
#define LDS_TOTAL 172548

enum { MODE_GELU_A1 = 0, MODE_ADD_XR = 1, MODE_SET_XR = 2, MODE_SET_XR_A0 = 3, MODE_ARGMIN = 4 };

// Force a device-memory load down the global_load_* path (AS 1), so flat_load
// never dual-counts DScnt against the LDS A-fragment traffic.
template <typename T>
__device__ __forceinline__ T gload(const T* p) {
  typedef const T __attribute__((address_space(1))) * gp_t;
  return *__builtin_bit_cast(gp_t, p);
}

__device__ __forceinline__ u16 f2bf(float f) {
  u32 u = __builtin_bit_cast(u32, f);
  u32 r = (u + 0x7FFFu + ((u >> 16) & 1u)) >> 16;
  return (u16)r;
}
__device__ __forceinline__ float bf2f(u16 b) {
  u32 u = ((u32)b) << 16;
  return __builtin_bit_cast(float, u);
}
__device__ __forceinline__ float gelu_f(float x) {
  return 0.5f * x * (1.0f + erff(x * 0.70710678118654752440f));
}
// float -> totally ordered key (ascending), packed with index.
__device__ __forceinline__ u64 packdi(float d, int i) {
  u32 u = __builtin_bit_cast(u32, d);
  u = (u & 0x80000000u) ? ~u : (u | 0x80000000u);
  return ((u64)u << 32) | (u32)i;
}
__device__ __forceinline__ v16bf frag(u32x4 a, u32x4 b) {
  union { u32x4 u[2]; v16bf v; } x;
  x.u[0] = a; x.u[1] = b;
  return x.v;
}

// One 16x16 C tile, K=Kd contraction. A row-major bf16 in LDS, B transposed
// bf16 [N][Kd] in global (L2-resident).
// A layout per ISA: lane half h gets K runs {8h..8h+7} and {16+8h..16+8h+7}.
// B layout per ISA: lane half h gets K runs {8h..8h+7} and {16+8h..16+8h+7}.
__device__ __forceinline__ v8f wmma_dot(const u16* __restrict__ arow,
                                        const u16* __restrict__ brow,
                                        int Kd, int hh) {
  v8f acc = {};
#pragma unroll 4
  for (int k0 = 0; k0 < Kd; k0 += 32) {
    u32x4 a0 = *(const u32x4*)(arow + k0 + hh * 8);
    u32x4 a1 = *(const u32x4*)(arow + k0 + 16 + hh * 8);
    u32x4 b0 = gload((const u32x4*)(brow + k0 + hh * 8));
    u32x4 b1 = gload((const u32x4*)(brow + k0 + 16 + hh * 8));
    acc = __builtin_amdgcn_wmma_f32_16x16x32_bf16(false, frag(a0, a1),
                                                  false, frag(b0, b1),
                                                  (short)0, acc, false, false);
  }
  return acc;
}

// Block-wide GEMM: C[32 x (16*Ntiles)] = A[32x512] @ Bt^T, with fused epilogue.
__device__ __forceinline__ void gemm512(const u16* __restrict__ A, const u16* __restrict__ Bg,
                                        int Ntiles, int mode, const float* __restrict__ bias,
                                        float* __restrict__ XR, u16* __restrict__ Aout,
                                        const float* __restrict__ cnorm, u64* __restrict__ best,
                                        int tid) {
  const int wave = tid >> 5, lane = tid & 31;
  const int l15 = lane & 15, hh = lane >> 4;
  if (mode == MODE_ARGMIN) {
    u64 lb[16];
#pragma unroll
    for (int s = 0; s < 16; ++s) lb[s] = ~0ull;
    for (int t = wave; t < 2 * Ntiles; t += 8) {
      const int mt = t & 1, nt = t >> 1;
      const int m = mt * 16 + l15, n = nt * 16 + l15;
      v8f acc = wmma_dot(A + m * 512, Bg + (size_t)n * 512, 512, hh);
      const float cn = gload(&cnorm[n]);
#pragma unroll
      for (int r = 0; r < 8; ++r) {
        float dist = cn - 2.0f * acc[r];   // + ||e||^2 is row-constant: argmin-invariant
        u64 p = packdi(dist, n);
        int s = mt * 8 + r;
        if (p < lb[s]) lb[s] = p;
      }
    }
#pragma unroll
    for (int s = 0; s < 16; ++s) {
      int mr = ((s >> 3) ? 16 : 0) + 8 * hh + (s & 7);
      if (mr < JJ && lb[s] != ~0ull) atomicMin(&best[mr], lb[s]);
    }
  } else {
    for (int t = wave; t < 2 * Ntiles; t += 8) {
      const int mt = t & 1, nt = t >> 1;
      const int m = mt * 16 + l15, n = nt * 16 + l15;
      v8f acc = wmma_dot(A + m * 512, Bg + (size_t)n * 512, 512, hh);
      const float bv = gload(&bias[n]);
#pragma unroll
      for (int r = 0; r < 8; ++r) {
        const int mr = mt * 16 + 8 * hh + r;
        float v = acc[r] + bv;
        if (mode == MODE_GELU_A1) {
          Aout[mr * 512 + n] = (mr < JJ) ? f2bf(gelu_f(v)) : (u16)0;
        } else if (mode == MODE_ADD_XR) {
          XR[mr * 512 + n] += v;
        } else if (mode == MODE_SET_XR) {
          XR[mr * 512 + n] = v;
        } else { // MODE_SET_XR_A0
          XR[mr * 512 + n] = v;
          Aout[mr * 512 + n] = (mr < JJ) ? f2bf(v) : (u16)0;
        }
      }
    }
  }
}

// LayerNorm row stats: one wave per row, shfl reduce (wave32).
__device__ __forceinline__ void ln_stats(const float* __restrict__ XR, float* __restrict__ mA,
                                         float* __restrict__ rA, int tid) {
  const int wave = tid >> 5, lane = tid & 31;
  for (int j = wave; j < JJ; j += 8) {
    float s = 0.f, s2 = 0.f;
    for (int k = lane; k < 512; k += 32) {
      float v = XR[j * 512 + k];
      s += v; s2 = fmaf(v, v, s2);
    }
    for (int off = 16; off; off >>= 1) {
      s  += __shfl_down(s,  off, 32);
      s2 += __shfl_down(s2, off, 32);
    }
    if (lane == 0) {
      float m = s * (1.0f / 512.0f);
      float var = s2 * (1.0f / 512.0f) - m * m;
      mA[j] = m;
      rA[j] = rsqrtf(var + 1e-5f);
    }
  }
}

// Apply LN (gamma/beta) -> row-major bf16 A buffer [JP][512]; pad rows zeroed.
__device__ __forceinline__ void ln_apply(const float* __restrict__ XR, u16* __restrict__ A,
                                         const float* __restrict__ g, const float* __restrict__ b,
                                         const float* __restrict__ mA, const float* __restrict__ rA,
                                         int tid) {
  for (int i = tid; i < JP * 512; i += THREADS) {
    int j = i >> 9, k = i & 511;
    float v = 0.f;
    if (j < JJ) v = (XR[i] - mA[j]) * rA[j] * gload(&g[k]) + gload(&b[k]);
    A[i] = f2bf(v);
  }
}

// Full mixer layer. Token MLP runs on WMMA via transposed staging:
//   AT[512][32pad] = LN1(x)^T ; T1 = gelu(AT @ tw1p + tb1) [512][64] ;
//   XR^T += T1 @ tw2p + tb2 ; then channel MLP on WMMA as before.
__device__ __forceinline__ void mixer(float* XR, u16* A0, u16* A1, u16* AT, u16* T1,
                                      float* mA, float* rA,
                                      const float* ln1g, const float* ln1b,
                                      const u16* tw1p, const float* tb1,
                                      const u16* tw2p, const float* tb2,
                                      const float* ln2g, const float* ln2b,
                                      const u16* cw1t, const float* cb1,
                                      const u16* cw2t, const float* cb2, int tid) {
  const int wave = tid >> 5, lane = tid & 31;
  const int l15 = lane & 15, hh = lane >> 4;

  // LN1 -> AT (transposed, K padded to 32 with zeros)
  ln_stats(XR, mA, rA, tid);
  __syncthreads();
  for (int i = tid; i < 512 * 32; i += THREADS) {
    int j = i >> 9, h = i & 511;   // coalesced XR read; 2-way-conflict AT write
    float v = 0.f;
    if (j < JJ) v = (XR[j * 512 + h] - mA[j]) * rA[j] * gload(&ln1g[h]) + gload(&ln1b[h]);
    AT[h * ATS + j] = f2bf(v);
  }
  __syncthreads();

  // token GEMM1: T1[512][64] = gelu(AT @ tw1p + tb1), K=32 (one WMMA per tile)
  for (int t = wave; t < 32 * 4; t += 8) {
    const int mt = t & 31, nt = t >> 5;
    const int m = mt * 16 + l15, n = nt * 16 + l15;
    v8f acc = wmma_dot(AT + m * ATS, tw1p + n * 32, 32, hh);
    const float bv = gload(&tb1[n]);
#pragma unroll
    for (int r = 0; r < 8; ++r) {
      const int mr = mt * 16 + 8 * hh + r;
      T1[mr * 64 + n] = f2bf(gelu_f(acc[r] + bv));
    }
  }
  __syncthreads();

  // token GEMM2: XR[j][h] += (T1 @ tw2p + tb2)[h][j], K=64, N=32 (23 valid)
  for (int t = wave; t < 32 * 2; t += 8) {
    const int mt = t & 31, nt = t >> 5;
    const int m = mt * 16 + l15, n = nt * 16 + l15;
    v8f acc = wmma_dot(T1 + m * 64, tw2p + n * 64, 64, hh);
    if (n < JJ) {
      const float bv = gload(&tb2[n]);
#pragma unroll
      for (int r = 0; r < 8; ++r) {
        const int mr = mt * 16 + 8 * hh + r;
        XR[n * 512 + mr] += acc[r] + bv;   // transposed residual add
      }
    }
  }
  __syncthreads();                          // XR = x + y

  // channel MLP (WMMA, K=512)
  ln_stats(XR, mA, rA, tid);
  __syncthreads();
  ln_apply(XR, A0, ln2g, ln2b, mA, rA, tid);
  __syncthreads();
  gemm512(A0, cw1t, NT_H, MODE_GELU_A1, cb1, nullptr, A1, nullptr, nullptr, tid);
  __syncthreads();
  gemm512(A1, cw2t, NT_H, MODE_ADD_XR, cb2, XR, nullptr, nullptr, nullptr, tid);
  __syncthreads();                          // XR = x + y + z
}

// ------------------------------- prep kernels ------------------------------

__global__ void k_tr_bf16(const float* __restrict__ src, u16* __restrict__ dst,
                          int Lc, int R, int C) {
  size_t total = (size_t)Lc * R * C;
  for (size_t i = (size_t)blockIdx.x * blockDim.x + threadIdx.x; i < total;
       i += (size_t)gridDim.x * blockDim.x) {
    size_t l = i / ((size_t)R * C);
    size_t rc = i - l * (size_t)R * C;
    int r = (int)(rc / C), c = (int)(rc % C);
    dst[l * (size_t)R * C + (size_t)c * R + r] = f2bf(src[i]);
  }
}

// src [Lc][K][N] -> dst [Lc][Np][Kp], zero-padded (B-native layout with pads).
__global__ void k_tr_pad(const float* __restrict__ src, u16* __restrict__ dst,
                         int Lc, int K, int N, int Kp, int Np) {
  size_t total = (size_t)Lc * Np * Kp;
  for (size_t i = (size_t)blockIdx.x * blockDim.x + threadIdx.x; i < total;
       i += (size_t)gridDim.x * blockDim.x) {
    size_t l = i / ((size_t)Np * Kp);
    size_t rem = i - l * (size_t)Np * Kp;
    int n = (int)(rem / Kp), k = (int)(rem % Kp);
    float v = (n < N && k < K) ? src[((size_t)l * K + k) * N + n] : 0.f;
    dst[i] = f2bf(v);
  }
}

__global__ void k_cvt_bf16(const float* __restrict__ src, u16* __restrict__ dst, int n) {
  for (int i = blockIdx.x * blockDim.x + threadIdx.x; i < n; i += gridDim.x * blockDim.x)
    dst[i] = f2bf(src[i]);
}

__global__ void k_cnorm(const float* __restrict__ cb, float* __restrict__ cn) {
  int k = blockIdx.x * blockDim.x + threadIdx.x;
  if (k < KK) {
    float s = 0.f;
    for (int d = 0; d < DD; ++d) {
      float v = cb[(size_t)k * DD + d];
      s = fmaf(v, v, s);
    }
    cn[k] = s;
  }
}

__global__ void k_zero1(float* p) { *p = 0.f; }

// ------------------------------ forward kernel -----------------------------

__global__ __launch_bounds__(THREADS)
void k_forward(const float* __restrict__ pose, const float* __restrict__ start_w,
               const float* __restrict__ start_b,
               const float* __restrict__ enc_ln1, const float* __restrict__ enc_tb1,
               const float* __restrict__ enc_tb2, const float* __restrict__ enc_ln2,
               const float* __restrict__ enc_cb1, const float* __restrict__ enc_cb2,
               const float* __restrict__ enc_lnf,
               const float* __restrict__ tok_w, const float* __restrict__ tok_b,
               const float* __restrict__ feat_b, const float* __restrict__ codebook,
               const float* __restrict__ dec_tok_w, const float* __restrict__ dec_tok_b,
               const float* __restrict__ dec_start_b,
               const float* __restrict__ dec_ln1, const float* __restrict__ dec_tb1,
               const float* __restrict__ dec_tb2, const float* __restrict__ dec_ln2,
               const float* __restrict__ dec_cb1, const float* __restrict__ dec_cb2,
               const float* __restrict__ dec_lnf,
               const float* __restrict__ rec_w, const float* __restrict__ rec_b,
               const u16* __restrict__ enc_cw1t, const u16* __restrict__ enc_cw2t,
               const u16* __restrict__ dec_cw1t, const u16* __restrict__ dec_cw2t,
               const u16* __restrict__ feat_wt, const u16* __restrict__ dec_start_wt,
               const u16* __restrict__ cb_bf, const float* __restrict__ cnorm,
               const u16* __restrict__ enc_tw1p, const u16* __restrict__ enc_tw2p,
               const u16* __restrict__ dec_tw1p, const u16* __restrict__ dec_tw2p,
               float* __restrict__ out_rec, float* __restrict__ out_idx,
               float* __restrict__ out_loss) {
  extern __shared__ unsigned char smem[];
  float* XR = (float*)(smem + LDS_XR);       // [JP][512] fp32 residual stream
  u16*   A0 = (u16*)(smem + LDS_A0);         // [JP][512] bf16 A-staging
  u16*   A1 = (u16*)(smem + LDS_A1);         // [JP][512] bf16 A-staging
  u16*   T1 = (u16*)(smem + LDS_A0);         // [512][64] bf16 token hidden (overlays A0+A1)
  u16*   AT = (u16*)(smem + LDS_AT);         // [512][ATS] bf16 transposed A
  float* mA = (float*)(smem + LDS_MEAN);
  float* rA = (float*)(smem + LDS_RSTD);
  u64* best = (u64*)(smem + LDS_BEST);
  float* lossAcc = (float*)(smem + LDS_LOSS);

  const int tid = threadIdx.x;
  const size_t s = blockIdx.x;

  // init: zero pad rows
  for (int i = tid; i < (JP - JJ) * 512; i += THREADS) {
    A0[JJ * 512 + i] = 0;
    A1[JJ * 512 + i] = 0;
    XR[JJ * 512 + i] = 0.f;
  }

  // x = pose @ start_w + start_b
  const float* ps = pose + s * JJ * 9;
  for (int i = tid; i < JJ * 512; i += THREADS) {
    int j = i >> 9, h = i & 511;
    float acc = gload(&start_b[h]);
#pragma unroll
    for (int c = 0; c < 9; ++c) acc = fmaf(gload(&ps[j * 9 + c]), gload(&start_w[c * 512 + h]), acc);
    XR[i] = acc;
  }
  __syncthreads();

  // encoder mixers
  for (int l = 0; l < LL; ++l)
    mixer(XR, A0, A1, AT, T1, mA, rA,
          enc_ln1 + l * 2 * HH, enc_ln1 + l * 2 * HH + HH,
          enc_tw1p + l * TIc * 32, enc_tb1 + l * TIc,
          enc_tw2p + l * 32 * TIc, enc_tb2 + l * JJ,
          enc_ln2 + l * 2 * HH, enc_ln2 + l * 2 * HH + HH,
          enc_cw1t + (size_t)l * HH * HH, enc_cb1 + l * HH,
          enc_cw2t + (size_t)l * HH * HH, enc_cb2 + l * HH, tid);

  // final encoder LN -> A0 (bf16)
  ln_stats(XR, mA, rA, tid); __syncthreads();
  ln_apply(XR, A0, enc_lnf, enc_lnf + HH, mA, rA, tid); __syncthreads();

  // token projection: xt = (xf^T @ tok_w + tok_b)^T  -> A1 (bf16)
  for (int h = tid; h < 512; h += THREADS) {
    float col[JJ];
    for (int j = 0; j < JJ; ++j) col[j] = bf2f(A0[j * 512 + h]);
    for (int j = 0; j < JJ; ++j) {
      float acc = tok_b[j];
      for (int j2 = 0; j2 < JJ; ++j2) acc = fmaf(col[j2], tok_w[j2 * JJ + j], acc);
      A1[j * 512 + h] = f2bf(acc);
    }
  }
  __syncthreads();

  // e = xt @ feat_w + feat_b  -> XR fp32 and A0 bf16
  gemm512(A1, feat_wt, NT_H, MODE_SET_XR_A0, feat_b, XR, A0, nullptr, nullptr, tid);
  if (tid < JP) best[tid] = ~0ull;
  if (tid == 0) *lossAcc = 0.f;
  __syncthreads();

  // VQ: dist partial = ||c||^2 - 2 e.c, argmin fused into WMMA epilogue
  gemm512(A0, cb_bf, NT_K, MODE_ARGMIN, nullptr, nullptr, nullptr, cnorm, best, tid);
  __syncthreads();

  // latent loss + quantize (q overwrites e in XR)
  float lsum = 0.f;
  for (int i = tid; i < JJ * 512; i += THREADS) {
    int j = i >> 9;
    int idx = (int)(u32)(best[j] & 0xffffffffull);
    float q = gload(&codebook[(size_t)idx * 512 + (i & 511)]);
    float e = XR[i];
    float d = q - e;
    lsum = fmaf(d, d, lsum);
    XR[i] = q;
  }
  atomicAdd(lossAcc, lsum);
  if (tid < JJ) out_idx[s * JJ + tid] = (float)(u32)(best[tid] & 0xffffffffull);
  __syncthreads();
  if (tid == 0) atomicAdd(out_loss, *lossAcc * (1.0f / ((float)BB * JJ * DD)));

  // decoder token projection -> A1 (bf16)
  for (int h = tid; h < 512; h += THREADS) {
    float col[JJ];
    for (int j = 0; j < JJ; ++j) col[j] = XR[j * 512 + h];
    for (int j = 0; j < JJ; ++j) {
      float acc = dec_tok_b[j];
      for (int j2 = 0; j2 < JJ; ++j2) acc = fmaf(col[j2], dec_tok_w[j2 * JJ + j], acc);
      A1[j * 512 + h] = f2bf(acc);
    }
  }
  __syncthreads();

  // y = t2 @ dec_start_w + dec_start_b -> XR
  gemm512(A1, dec_start_wt, NT_H, MODE_SET_XR, dec_start_b, XR, nullptr, nullptr, nullptr, tid);
  __syncthreads();

  // decoder mixers
  for (int l = 0; l < LL; ++l)
    mixer(XR, A0, A1, AT, T1, mA, rA,
          dec_ln1 + l * 2 * HH, dec_ln1 + l * 2 * HH + HH,
          dec_tw1p + l * TIc * 32, dec_tb1 + l * TIc,
          dec_tw2p + l * 32 * TIc, dec_tb2 + l * JJ,
          dec_ln2 + l * 2 * HH, dec_ln2 + l * 2 * HH + HH,
          dec_cw1t + (size_t)l * HH * HH, dec_cb1 + l * HH,
          dec_cw2t + (size_t)l * HH * HH, dec_cb2 + l * HH, tid);

  // final LN + recovered = lnf(y) @ rec_w + rec_b
  ln_stats(XR, mA, rA, tid); __syncthreads();
  for (int i = tid; i < JJ * 9; i += THREADS) {
    int j = i / 9, c = i % 9;
    float m = mA[j], rs = rA[j];
    float acc = gload(&rec_b[c]);
    for (int h = 0; h < 512; ++h) {
      float xf = (XR[j * 512 + h] - m) * rs * gload(&dec_lnf[h]) + gload(&dec_lnf[512 + h]);
      acc = fmaf(xf, gload(&rec_w[h * 9 + c]), acc);
    }
    out_rec[s * (JJ * 9) + i] = acc;
  }
}

// --------------------------------- launch ----------------------------------

extern "C" void kernel_launch(void* const* d_in, const int* in_sizes, int n_in,
                              void* d_out, int out_size, void* d_ws, size_t ws_size,
                              hipStream_t stream) {
  (void)in_sizes; (void)n_in; (void)out_size; (void)ws_size;
  const float* pose        = (const float*)d_in[0];
  const float* start_w     = (const float*)d_in[1];
  const float* start_b     = (const float*)d_in[2];
  const float* enc_ln1     = (const float*)d_in[3];
  const float* enc_tw1     = (const float*)d_in[4];
  const float* enc_tb1     = (const float*)d_in[5];
  const float* enc_tw2     = (const float*)d_in[6];
  const float* enc_tb2     = (const float*)d_in[7];
  const float* enc_ln2     = (const float*)d_in[8];
  const float* enc_cw1     = (const float*)d_in[9];
  const float* enc_cb1     = (const float*)d_in[10];
  const float* enc_cw2     = (const float*)d_in[11];
  const float* enc_cb2     = (const float*)d_in[12];
  const float* enc_lnf     = (const float*)d_in[13];
  const float* tok_w       = (const float*)d_in[14];
  const float* tok_b       = (const float*)d_in[15];
  const float* feat_w      = (const float*)d_in[16];
  const float* feat_b      = (const float*)d_in[17];
  const float* codebook    = (const float*)d_in[18];
  const float* dec_tok_w   = (const float*)d_in[19];
  const float* dec_tok_b   = (const float*)d_in[20];
  const float* dec_start_w = (const float*)d_in[21];
  const float* dec_start_b = (const float*)d_in[22];
  const float* dec_ln1     = (const float*)d_in[23];
  const float* dec_tw1     = (const float*)d_in[24];
  const float* dec_tb1     = (const float*)d_in[25];
  const float* dec_tw2     = (const float*)d_in[26];
  const float* dec_tb2     = (const float*)d_in[27];
  const float* dec_ln2     = (const float*)d_in[28];
  const float* dec_cw1     = (const float*)d_in[29];
  const float* dec_cb1     = (const float*)d_in[30];
  const float* dec_cw2     = (const float*)d_in[31];
  const float* dec_cb2     = (const float*)d_in[32];
  const float* dec_lnf     = (const float*)d_in[33];
  const float* rec_w       = (const float*)d_in[34];
  const float* rec_b       = (const float*)d_in[35];

  // workspace layout
  unsigned char* ws = (unsigned char*)d_ws;
  size_t o = 0;
  float* cnorm        = (float*)(ws + o); o += (size_t)KK * 4;
  u16* enc_cw1t       = (u16*)(ws + o);   o += (size_t)LL * HH * HH * 2;
  u16* enc_cw2t       = (u16*)(ws + o);   o += (size_t)LL * HH * HH * 2;
  u16* dec_cw1t       = (u16*)(ws + o);   o += (size_t)LL * HH * HH * 2;
  u16* dec_cw2t       = (u16*)(ws + o);   o += (size_t)LL * HH * HH * 2;
  u16* feat_wt        = (u16*)(ws + o);   o += (size_t)HH * DD * 2;
  u16* dec_start_wt   = (u16*)(ws + o);   o += (size_t)DD * HH * 2;
  u16* cb_bf          = (u16*)(ws + o);   o += (size_t)KK * DD * 2;
  u16* enc_tw1p       = (u16*)(ws + o);   o += (size_t)LL * TIc * 32 * 2;
  u16* enc_tw2p       = (u16*)(ws + o);   o += (size_t)LL * 32 * TIc * 2;
  u16* dec_tw1p       = (u16*)(ws + o);   o += (size_t)LL * TIc * 32 * 2;
  u16* dec_tw2p       = (u16*)(ws + o);   o += (size_t)LL * 32 * TIc * 2;

  float* out = (float*)d_out;
  float* out_rec  = out;                             // [B*J*9]
  float* out_idx  = out + (size_t)BB * JJ * 9;       // [B*J]
  float* out_loss = out_idx + (size_t)BB * JJ;       // [1]

  // weight prep (recomputed each launch: deterministic, graph-capture safe)
  k_tr_bf16<<<512, THREADS, 0, stream>>>(enc_cw1, enc_cw1t, LL, HH, HH);
  k_tr_bf16<<<512, THREADS, 0, stream>>>(enc_cw2, enc_cw2t, LL, HH, HH);
  k_tr_bf16<<<512, THREADS, 0, stream>>>(dec_cw1, dec_cw1t, LL, HH, HH);
  k_tr_bf16<<<512, THREADS, 0, stream>>>(dec_cw2, dec_cw2t, LL, HH, HH);
  k_tr_bf16<<<256, THREADS, 0, stream>>>(feat_w, feat_wt, 1, HH, DD);
  k_tr_bf16<<<256, THREADS, 0, stream>>>(dec_start_w, dec_start_wt, 1, DD, HH);
  k_cvt_bf16<<<512, THREADS, 0, stream>>>(codebook, cb_bf, KK * DD);
  k_cnorm<<<KK / THREADS, THREADS, 0, stream>>>(codebook, cnorm);
  // token weights -> zero-padded B-native layouts
  k_tr_pad<<<32, THREADS, 0, stream>>>(enc_tw1, enc_tw1p, LL, JJ, TIc, 32, TIc);
  k_tr_pad<<<32, THREADS, 0, stream>>>(enc_tw2, enc_tw2p, LL, TIc, JJ, TIc, 32);
  k_tr_pad<<<32, THREADS, 0, stream>>>(dec_tw1, dec_tw1p, LL, JJ, TIc, 32, TIc);
  k_tr_pad<<<32, THREADS, 0, stream>>>(dec_tw2, dec_tw2p, LL, TIc, JJ, TIc, 32);
  k_zero1<<<1, 1, 0, stream>>>(out_loss);

  (void)hipFuncSetAttribute((const void*)k_forward,
                            hipFuncAttributeMaxDynamicSharedMemorySize, LDS_TOTAL);
  k_forward<<<BB, THREADS, LDS_TOTAL, stream>>>(
      pose, start_w, start_b,
      enc_ln1, enc_tb1, enc_tb2, enc_ln2, enc_cb1, enc_cb2, enc_lnf,
      tok_w, tok_b, feat_b, codebook, dec_tok_w, dec_tok_b, dec_start_b,
      dec_ln1, dec_tb1, dec_tb2, dec_ln2, dec_cb1, dec_cb2, dec_lnf,
      rec_w, rec_b,
      enc_cw1t, enc_cw2t, dec_cw1t, dec_cw2t, feat_wt, dec_start_wt, cb_bf, cnorm,
      enc_tw1p, enc_tw2p, dec_tw1p, dec_tw2p,
      out_rec, out_idx, out_loss);
}
